// GIN_43490838839750
// MI455X (gfx1250) — matrix operand order repaired and checked by reading, not compiled, not run on hardware
//
#include <hip/hip_runtime.h>
#include <hip/hip_bf16.h>

typedef __attribute__((ext_vector_type(16))) _Float16 v16h;
typedef __attribute__((ext_vector_type(8)))  _Float16 v8h;
typedef __attribute__((ext_vector_type(8)))  float    v8f;

#define DEV static __device__ __forceinline__

// A-matrix fragment (16x32 f16): lane l holds row m=l&15;
// VGPR0-3 = K = sel*8 .. sel*8+7, VGPR4-7 = K = 16+sel*8 .. (sel = l>>4)
DEV v16h frag_a(const _Float16* base, int ld, int lane) {
  const int m = lane & 15, sel = (lane >> 4) & 1;
  const _Float16* p = base + m * ld + sel * 8;
  v8h lo = *(const v8h*)(p);
  v8h hi = *(const v8h*)(p + 16);
  v16h r;
#pragma unroll
  for (int i = 0; i < 8; ++i) { r[i] = lo[i]; r[i + 8] = hi[i]; }
  return r;
}

// B-matrix fragment (32x16 f16) from a source stored transposed as [col][K]:
// lane l holds col n=l&15, K = sel*8 .. sel*8+7, then 16+sel*8 .. 16+sel*8+7.
DEV v16h frag_b(const _Float16* base, int ld, int lane) {
  const int n = lane & 15, sel = (lane >> 4) & 1;
  const _Float16* p = base + n * ld + sel * 8;
  v8h lo = *(const v8h*)(p);
  v8h hi = *(const v8h*)(p + 16);
  v16h r;
#pragma unroll
  for (int i = 0; i < 8; ++i) { r[i] = lo[i]; r[i + 8] = hi[i]; }
  return r;
}

DEV v8f wmma16(v16h a, v16h b, v8f c) {
  return __builtin_amdgcn_wmma_f32_16x16x32_f16(false, a, false, b, (short)0, c,
                                                false, false);
}

// ---- d[i] = rsqrt(sum_j(adj[i,j]) + 1) ----
__global__ void k_rowsum_d(const float* __restrict__ adj, float* __restrict__ d) {
  int wave = threadIdx.x >> 5, lane = threadIdx.x & 31;
  int i = blockIdx.x * 8 + wave;
  float s = 0.f;
  for (int j = lane; j < 512; j += 32) s += adj[i * 512 + j];
#pragma unroll
  for (int m = 16; m >= 1; m >>= 1) s += __shfl_xor(s, m, 32);
  if (lane == 0) d[i] = rsqrtf(s + 1.0f);
}

// ---- noradj[i,j] = d[i]*(adj+I)[i,j]*d[j] (f32 + f16 copies) ----
__global__ void k_noradj(const float* __restrict__ adj, const float* __restrict__ d,
                         float* __restrict__ noradj, _Float16* __restrict__ nr16) {
  int i = blockIdx.x;
  int j = blockIdx.y * 128 + threadIdx.x;
  float v = adj[i * 512 + j] + (i == j ? 1.f : 0.f);
  v *= d[i] * d[j];
  noradj[i * 512 + j] = v;
  nr16[i * 512 + j] = (_Float16)v;
}

// ---- c[j] = sum_i noradj[i,j] ----
__global__ void k_colsum(const float* __restrict__ noradj, float* __restrict__ c) {
  int j = blockIdx.x * 64 + threadIdx.x;
  float s = 0.f;
  for (int i = 0; i < 512; ++i) s += noradj[i * 512 + j];
  c[j] = s;
}

__global__ void k_cvt_f16(const float* __restrict__ in, _Float16* __restrict__ out, int n) {
  int i = blockIdx.x * 256 + threadIdx.x;
  if (i < n) out[i] = (_Float16)in[i];
}

// out[n*K + k] = (f16) in[k*Ncol + n]   (K x Ncol row-major -> Ncol x K)
__global__ void k_transpose_f16(const float* __restrict__ in, _Float16* __restrict__ out,
                                int K, int Ncol) {
  int idx = blockIdx.x * 256 + threadIdx.x;
  if (idx < K * Ncol) {
    int n = idx / K, k = idx % K;
    out[idx] = (_Float16)in[k * Ncol + n];
  }
}

// ---- projections: [2048x64] @ [64x256] for q (scaled 1/8) and k, head-major out ----
__global__ void k_proj(const _Float16* __restrict__ xh, const _Float16* __restrict__ WqT,
                       const _Float16* __restrict__ WkT, const float* __restrict__ bq,
                       const float* __restrict__ bk, _Float16* __restrict__ q16,
                       _Float16* __restrict__ k16) {
  int lane = threadIdx.x;
  int mt = blockIdx.x * 16;
  int y = blockIdx.y;
  bool isK = y >= 16;
  int nt = (y & 15) * 16;
  const _Float16* WT = isK ? WkT : WqT;
  const float* bias = isK ? bk : bq;
  _Float16* dst = isK ? k16 : q16;
  float scale = isK ? 1.f : 0.125f;  // fold 1/sqrt(F) into q
  v8f acc = {};
#pragma unroll
  for (int ks = 0; ks < 64; ks += 32) {
    v16h a = frag_a(xh + mt * 64 + ks, 64, lane);
    v16h b = frag_b(WT + nt * 64 + ks, 64, lane);
    acc = wmma16(a, b, acc);
  }
  int n = lane & 15, sel = lane >> 4;
  int col = nt + n, h = col >> 6, f = col & 63;
#pragma unroll
  for (int r = 0; r < 8; ++r) {
    int row = mt + r + 8 * sel;
    int b_ = row >> 9, nn = row & 511;
    dst[(((b_ * 4 + h) * 512) + nn) * 64 + f] =
        (_Float16)((acc[r] + bias[col]) * scale);
  }
}

// ---- scores: per (b,h), S = q @ k^T (scale already in q) ----
__global__ void k_scores(const _Float16* __restrict__ q16, const _Float16* __restrict__ k16,
                         float* __restrict__ S) {
  int lane = threadIdx.x & 31, w = threadIdx.x >> 5;
  int bh = blockIdx.x;
  int it = blockIdx.y * 64 + (w >> 1) * 32;
  int jt = blockIdx.z * 64 + (w & 1) * 32;
  const _Float16* qb = q16 + bh * 512 * 64;
  const _Float16* kb = k16 + bh * 512 * 64;
  v8f acc[2][2] = {};
#pragma unroll
  for (int ks = 0; ks < 64; ks += 32) {
    v16h a0 = frag_a(qb + it * 64 + ks, 64, lane);
    v16h a1 = frag_a(qb + (it + 16) * 64 + ks, 64, lane);
    v16h b0 = frag_b(kb + jt * 64 + ks, 64, lane);      // k^T: contiguous rows of k
    v16h b1 = frag_b(kb + (jt + 16) * 64 + ks, 64, lane);
    acc[0][0] = wmma16(a0, b0, acc[0][0]);
    acc[0][1] = wmma16(a0, b1, acc[0][1]);
    acc[1][0] = wmma16(a1, b0, acc[1][0]);
    acc[1][1] = wmma16(a1, b1, acc[1][1]);
  }
  float* Sb = S + bh * 512 * 512;
  int n = lane & 15, sel = lane >> 4;
#pragma unroll
  for (int mi = 0; mi < 2; ++mi)
#pragma unroll
    for (int ni = 0; ni < 2; ++ni)
#pragma unroll
      for (int r = 0; r < 8; ++r)
        Sb[(it + mi * 16 + r + 8 * sel) * 512 + jt + ni * 16 + n] = acc[mi][ni][r];
}

// ---- in-place row softmax over 512 cols; one wave per row ----
__global__ void k_softmax(float* __restrict__ S) {
  int lane = threadIdx.x & 31, g = threadIdx.x >> 5;
  int row = blockIdx.x * 4 + g;
  float* p = S + (size_t)row * 512;
  float v[16];
  float m = -3.0e38f;
#pragma unroll
  for (int t = 0; t < 16; ++t) { v[t] = p[lane + t * 32]; m = fmaxf(m, v[t]); }
#pragma unroll
  for (int s = 16; s >= 1; s >>= 1) m = fmaxf(m, __shfl_xor(m, s, 32));
  float z = 0.f;
#pragma unroll
  for (int t = 0; t < 16; ++t) { v[t] = __expf(v[t] - m); z += v[t]; }
#pragma unroll
  for (int s = 16; s >= 1; s >>= 1) z += __shfl_xor(z, s, 32);
  float inv = 1.f / z;
#pragma unroll
  for (int t = 0; t < 16; ++t) p[lane + t * 32] = v[t] * inv;
}

// ---- T[b,h,j] = sum_i noradj[i,j] * attn[b,h,i,j] ----
__global__ void k_T(const float* __restrict__ S, const float* __restrict__ noradj,
                    float* __restrict__ T) {
  int bh = blockIdx.x;
  int j = blockIdx.y * 64 + threadIdx.x;
  const float* P = S + (size_t)bh * 512 * 512;
  float acc = 0.f;
  for (int i = 0; i < 512; ++i) acc += noradj[i * 512 + j] * P[i * 512 + j];
  T[bh * 512 + j] = acc;
}

// ---- sxT[b][f][j] = (f16)( x[b,j,f] * (sum_h T[b,h,j]*Wlin[h,f] + blin[f]*c[j]) ) ----
__global__ void k_sx(const float* __restrict__ T, const float* __restrict__ Wlin,
                     const float* __restrict__ blin, const float* __restrict__ c,
                     const float* __restrict__ x, _Float16* __restrict__ sxT) {
  int f = threadIdx.x;    // 64
  int j = blockIdx.y;     // 512
  int b = blockIdx.x;     // 4
  float s = blin[f] * c[j];
#pragma unroll
  for (int h = 0; h < 4; ++h) s += T[((b * 4 + h) * 512) + j] * Wlin[h * 64 + f];
  float sx = x[(b * 512 + j) * 64 + f] * s;
  sxT[(b * 64 + f) * 512 + j] = (_Float16)sx;
}

// ---- out = relu( (noradj @ sx) @ Wfc + bfc ), fused two WMMA GEMMs ----
__global__ void k_out(const _Float16* __restrict__ nr16, const _Float16* __restrict__ sxT,
                      const _Float16* __restrict__ WfcT, const float* __restrict__ bfc,
                      float* __restrict__ out) {
  __shared__ _Float16 lds[16 * 64];
  int lane = threadIdx.x;
  int b = blockIdx.x, it = blockIdx.y * 16;
  v8f acc[4] = {};
  for (int kt = 0; kt < 512; kt += 32) {
    v16h a = frag_a(nr16 + it * 512 + kt, 512, lane);
#pragma unroll
    for (int n = 0; n < 4; ++n) {
      v16h bb = frag_b(sxT + b * 64 * 512 + n * 16 * 512 + kt, 512, lane);
      acc[n] = wmma16(a, bb, acc[n]);
    }
  }
  int n15 = lane & 15, sel = lane >> 4;
  // bounce out1 (16x64) through LDS to re-shape C-layout -> A-layout in f16
#pragma unroll
  for (int nn = 0; nn < 4; ++nn)
#pragma unroll
    for (int r = 0; r < 8; ++r)
      lds[(r + 8 * sel) * 64 + nn * 16 + n15] = (_Float16)acc[nn][r];
  __syncthreads();
  v8f acc2[4] = {};
#pragma unroll
  for (int ks = 0; ks < 64; ks += 32) {
    v16h a = frag_a(lds + ks, 64, lane);
#pragma unroll
    for (int nn = 0; nn < 4; ++nn) {
      v16h bb = frag_b(WfcT + nn * 16 * 64 + ks, 64, lane);
      acc2[nn] = wmma16(a, bb, acc2[nn]);
    }
  }
#pragma unroll
  for (int nn = 0; nn < 4; ++nn) {
    int col = nn * 16 + n15;
#pragma unroll
    for (int r = 0; r < 8; ++r) {
      float v = acc2[nn][r] + bfc[col];
      out[((b * 512) + it + r + 8 * sel) * 64 + col] = fmaxf(v, 0.f);
    }
  }
}

extern "C" void kernel_launch(void* const* d_in, const int* in_sizes, int n_in,
                              void* d_out, int out_size, void* d_ws, size_t ws_size,
                              hipStream_t stream) {
  const float* x    = (const float*)d_in[0];
  const float* adj  = (const float*)d_in[1];
  const float* Wq   = (const float*)d_in[2];
  const float* bq   = (const float*)d_in[3];
  const float* Wk   = (const float*)d_in[4];
  const float* bk   = (const float*)d_in[5];
  const float* Wlin = (const float*)d_in[6];
  const float* blin = (const float*)d_in[7];
  const float* Wfc  = (const float*)d_in[8];
  const float* bfc  = (const float*)d_in[9];
  float* out = (float*)d_out;
  (void)in_sizes; (void)n_in; (void)out_size; (void)ws_size;

  char* ws = (char*)d_ws;
  size_t off = 0;
  auto alloc = [&](size_t bytes) {
    size_t o = off;
    off += (bytes + 255) & ~size_t(255);
    return o;
  };
  float*    S      = (float*)(ws + alloc((size_t)16 * 512 * 512 * 4)); // scores/attn
  _Float16* q16    = (_Float16*)(ws + alloc((size_t)16 * 512 * 64 * 2));
  _Float16* k16    = (_Float16*)(ws + alloc((size_t)16 * 512 * 64 * 2));
  _Float16* xh     = (_Float16*)(ws + alloc((size_t)2048 * 64 * 2));
  _Float16* WqT    = (_Float16*)(ws + alloc((size_t)256 * 64 * 2));
  _Float16* WkT    = (_Float16*)(ws + alloc((size_t)256 * 64 * 2));
  _Float16* WfcT   = (_Float16*)(ws + alloc((size_t)64 * 64 * 2));
  float*    noradj = (float*)(ws + alloc((size_t)512 * 512 * 4));
  _Float16* nr16   = (_Float16*)(ws + alloc((size_t)512 * 512 * 2));
  float*    dvec   = (float*)(ws + alloc(512 * 4));
  float*    cvec   = (float*)(ws + alloc(512 * 4));
  float*    Tbuf   = (float*)(ws + alloc((size_t)16 * 512 * 4));
  _Float16* sxT    = (_Float16*)(ws + alloc((size_t)4 * 64 * 512 * 2));

  k_rowsum_d<<<64, 256, 0, stream>>>(adj, dvec);
  k_noradj<<<dim3(512, 4), 128, 0, stream>>>(adj, dvec, noradj, nr16);
  k_colsum<<<8, 64, 0, stream>>>(noradj, cvec);
  k_cvt_f16<<<(2048 * 64 + 255) / 256, 256, 0, stream>>>(x, xh, 2048 * 64);
  k_transpose_f16<<<(64 * 256 + 255) / 256, 256, 0, stream>>>(Wq, WqT, 64, 256);
  k_transpose_f16<<<(64 * 256 + 255) / 256, 256, 0, stream>>>(Wk, WkT, 64, 256);
  k_transpose_f16<<<(64 * 64 + 255) / 256, 256, 0, stream>>>(Wfc, WfcT, 64, 64);
  k_proj<<<dim3(128, 32), 32, 0, stream>>>(xh, WqT, WkT, bq, bk, q16, k16);
  k_scores<<<dim3(16, 8, 8), 128, 0, stream>>>(q16, k16, S);
  k_softmax<<<2048, 128, 0, stream>>>(S);
  k_T<<<dim3(16, 8), 64, 0, stream>>>(S, noradj, Tbuf);
  k_sx<<<dim3(4, 512), 64, 0, stream>>>(Tbuf, Wlin, blin, cvec, x, sxT);
  k_out<<<dim3(4, 32), 32, 0, stream>>>(nr16, sxT, WfcT, bfc, out);
}
